// MoE_78151224918194
// MI455X (gfx1250) — matrix operand order repaired and checked by reading, not compiled, not run on hardware
//
#include <hip/hip_runtime.h>
#include <hip/hip_bf16.h>
#include <cstdint>
#include <cstddef>

// Problem constants (match reference)
#define H_DIM  2048
#define I_DIM  1408
#define E_NUM  8
#define IS_DIM 2816
#define T_NUM  8192   // 4 * 2048 tokens

typedef __bf16 bf16;
typedef __attribute__((ext_vector_type(16))) __bf16 v16bf;
typedef __attribute__((ext_vector_type(8)))  float  v8f;

union FragBF { uint4 u[2]; v16bf v; };   // 32 bytes
union Pack8  { bf16 h[8]; uint4 u; };    // 16 bytes

#define BK 64
#define LDS_STRIDE 72   // 64 + 8 pad (bf16 elems): 144B rows, 16B aligned
#define TILE_ELEMS (128 * LDS_STRIDE)

// --------------------------------------------------------------------------
// CDNA5 async global->LDS copy (ASYNCcnt-tracked), with portable fallback.
// Builtin signature (from hipcc diagnostic): (v4i AS1*, v4i AS3*, Imm, Imm)
// --------------------------------------------------------------------------
#if __has_builtin(__builtin_amdgcn_global_load_async_to_lds_b128) && \
    __has_builtin(__builtin_amdgcn_s_wait_asynccnt)
#define USE_ASYNC_LDS 1
#else
#define USE_ASYNC_LDS 0
#endif

typedef int v4i __attribute__((vector_size(16)));
typedef __attribute__((address_space(1))) v4i* gptr_v4i;
typedef __attribute__((address_space(3))) v4i* lptr_v4i;

__device__ __forceinline__ void async_copy16(const bf16* g, bf16* l) {
#if USE_ASYNC_LDS
  __builtin_amdgcn_global_load_async_to_lds_b128(
      (gptr_v4i)(void*)g, (lptr_v4i)(void*)l, 0, 0);
#else
  *(uint4*)l = *(const uint4*)g;
#endif
}
__device__ __forceinline__ void async_wait_all() {
#if USE_ASYNC_LDS
  __builtin_amdgcn_s_wait_asynccnt(0);
#endif
}

// ---------------------------------------------------------------------------
// fp32 -> bf16 convert of hidden states (8 elements / thread)
// ---------------------------------------------------------------------------
__global__ __launch_bounds__(256)
void moe_convert_x(const float* __restrict__ X, bf16* __restrict__ Xb) {
  size_t i = ((size_t)blockIdx.x * 256 + threadIdx.x) * 8;
  float4 a = *(const float4*)(X + i);
  float4 b = *(const float4*)(X + i + 4);
  Pack8 p;
  p.h[0] = (bf16)a.x; p.h[1] = (bf16)a.y; p.h[2] = (bf16)a.z; p.h[3] = (bf16)a.w;
  p.h[4] = (bf16)b.x; p.h[5] = (bf16)b.y; p.h[6] = (bf16)b.z; p.h[7] = (bf16)b.w;
  *(uint4*)(Xb + i) = p.u;
}

// ---------------------------------------------------------------------------
// Gate: logits (x @ gate_w^T), softmax over 8, top-2, renorm, append to
// per-expert token lists. One wave (32 lanes) per token, 8 tokens per block.
// ---------------------------------------------------------------------------
__global__ __launch_bounds__(256)
void moe_gate_route(const float* __restrict__ X, const float* __restrict__ GW,
                    int* __restrict__ counts, int* __restrict__ elist,
                    float* __restrict__ wlist) {
  const int lane = threadIdx.x & 31;
  const int wave = threadIdx.x >> 5;
  const int t = blockIdx.x * 8 + wave;

  float acc[E_NUM];
#pragma unroll
  for (int e = 0; e < E_NUM; ++e) acc[e] = 0.0f;

  const float* xr = X + (size_t)t * H_DIM;
  for (int k = lane; k < H_DIM; k += 32) {
    float xv = xr[k];
#pragma unroll
    for (int e = 0; e < E_NUM; ++e) acc[e] += xv * GW[e * H_DIM + k];
  }
#pragma unroll
  for (int off = 16; off > 0; off >>= 1) {
#pragma unroll
    for (int e = 0; e < E_NUM; ++e) acc[e] += __shfl_xor(acc[e], off, 32);
  }

  if (lane == 0) {
    float m = acc[0];
#pragma unroll
    for (int e = 1; e < E_NUM; ++e) m = fmaxf(m, acc[e]);
    float p[E_NUM], s = 0.0f;
#pragma unroll
    for (int e = 0; e < E_NUM; ++e) { p[e] = __expf(acc[e] - m); s += p[e]; }
    float inv_s = 1.0f / s;
#pragma unroll
    for (int e = 0; e < E_NUM; ++e) p[e] *= inv_s;

    int i1 = 0;
#pragma unroll
    for (int e = 1; e < E_NUM; ++e) if (p[e] > p[i1]) i1 = e;
    int i2 = (i1 == 0) ? 1 : 0;
#pragma unroll
    for (int e = 0; e < E_NUM; ++e) if (e != i1 && p[e] > p[i2]) i2 = e;

    float w1 = p[i1], w2 = p[i2];
    float invn = 1.0f / (w1 + w2 + 1e-20f);
    w1 *= invn; w2 *= invn;

    int s1 = atomicAdd(&counts[i1], 1);
    elist[(size_t)i1 * T_NUM + s1] = t;
    wlist[(size_t)i1 * T_NUM + s1] = w1;
    int s2 = atomicAdd(&counts[i2], 1);
    elist[(size_t)i2 * T_NUM + s2] = t;
    wlist[(size_t)i2 * T_NUM + s2] = w2;
  }
}

__global__ void moe_prefix8(const int* __restrict__ counts, int* __restrict__ offsets) {
  if (threadIdx.x == 0) {
    int s = 0;
#pragma unroll
    for (int e = 0; e < E_NUM; ++e) { offsets[e] = s; s += counts[e]; }
  }
}

// ---------------------------------------------------------------------------
// Fragment loaders (LDS -> VGPR, §7.12.2 layouts)
// A in LDS: row-major [128][LDS_STRIDE] bf16 (M x K)
// B in LDS: [128][LDS_STRIDE] bf16, N-major (transposed: row = N, col = K)
// ---------------------------------------------------------------------------
__device__ __forceinline__ v16bf load_a_frag(const bf16* As, int row0, int ks, int lane) {
  const int r  = row0 + (lane & 15);
  const int k0 = ks + ((lane & 16) ? 8 : 0);
  FragBF f;
  f.u[0] = *(const uint4*)(As + r * LDS_STRIDE + k0);       // K = k0..k0+7
  f.u[1] = *(const uint4*)(As + r * LDS_STRIDE + k0 + 16);  // K = k0+16..k0+23
  return f.v;
}
__device__ __forceinline__ v16bf load_b_frag(const bf16* Bs, int col0, int ks, int lane) {
  const int n  = col0 + (lane & 15);
  const int k0 = ks + ((lane & 16) ? 8 : 0);
  FragBF f;
  f.u[0] = *(const uint4*)(Bs + n * LDS_STRIDE + k0);       // K = k0..k0+7
  f.u[1] = *(const uint4*)(Bs + n * LDS_STRIDE + k0 + 16);  // K = k0+16..k0+23
  return f.v;
}

// pick float4 component by unrolled constant index
__device__ __forceinline__ float f4c(const float4& v, int j) {
  return j == 0 ? v.x : j == 1 ? v.y : j == 2 ? v.z : v.w;
}

// ---------------------------------------------------------------------------
// GEMM1: Act = silu(X @ Wg) * (X @ Wu), bf16 out. 128x128 block tile,
// 8 waves of 64(M)x32(N), K-staged by 64 with double-buffered LDS:
//  - A tile copied global->LDS via async b128 copies (ASYNCcnt)
//  - B tiles staged through registers k-major, packed ds_store_b128
// routed mode (counts != null): rows gathered via elist, output compacted.
// ---------------------------------------------------------------------------
__global__ __launch_bounds__(256)
void moe_gemm1_swiglu(const bf16* __restrict__ X,   // [T][H_DIM] bf16
                      const float* __restrict__ Wg, // [K][N] fp32 (per expert K*N)
                      const float* __restrict__ Wu,
                      bf16* __restrict__ Act,       // [rows][N] bf16
                      int K, int N,
                      const int* __restrict__ counts,
                      const int* __restrict__ offsets,
                      const int* __restrict__ elist) {
  __shared__ bf16 As [2][TILE_ELEMS];
  __shared__ bf16 Bgs[2][TILE_ELEMS];
  __shared__ bf16 Bus[2][TILE_ELEMS];
  __shared__ int  tokLds[128];

  const int tid  = threadIdx.x;
  const int lane = tid & 31;
  const int wave = tid >> 5;
  const int mtile = blockIdx.x, ntile = blockIdx.y, e = blockIdx.z;
  const bool routed = (counts != nullptr);

  int cnt = T_NUM, rowOff = 0;
  if (routed) {
    cnt = counts[e];
    if (mtile * 128 >= cnt) return;      // uniform early exit
    rowOff = offsets[e];
    Wg += (size_t)e * K * N;
    Wu += (size_t)e * K * N;
  }
  const int mbase = mtile * 128;
  const int nbase = ntile * 128;

  if (routed && tid < 128) {
    int pos = mbase + tid;
    tokLds[tid] = (pos < cnt) ? elist[(size_t)e * T_NUM + pos] : 0;
  }
  __syncthreads();   // tokLds visible to all fill issues

  const int wm = (wave & 1) * 64;
  const int wn = (wave >> 1) * 32;
  const int nB = (tid & 31) * 4;     // B-fill: column group
  const int kB = (tid >> 5) * 8;     // B-fill: k group (8 consecutive K)

  float4 rg[8], ru[8];               // B staging registers

  v8f cg[4][2], cu[4][2];
#pragma unroll
  for (int i = 0; i < 4; ++i)
#pragma unroll
    for (int j = 0; j < 2; ++j) { cg[i][j] = {}; cu[i][j] = {}; }

  // ---- fill helpers ----
  auto fillA = [&](int buf, int kb) {
#pragma unroll
    for (int i = 0; i < 4; ++i) {
      int idx = tid + i * 256;       // 0..1023
      int r = idx >> 3;
      int c = (idx & 7) * 8;
      size_t row = routed ? (size_t)tokLds[r] : (size_t)(mbase + r);
      async_copy16(X + row * H_DIM + kb + c, &As[buf][r * LDS_STRIDE + c]);
    }
  };
  auto loadB = [&](int kb) {
#pragma unroll
    for (int kk = 0; kk < 8; ++kk) {
      size_t off = (size_t)(kb + kB + kk) * N + nbase + nB;
      rg[kk] = *(const float4*)(Wg + off);
      ru[kk] = *(const float4*)(Wu + off);
    }
  };
  auto storeB = [&](int buf) {
#pragma unroll
    for (int j = 0; j < 4; ++j) {
      Pack8 pg, pu;
#pragma unroll
      for (int kk = 0; kk < 8; ++kk) {
        pg.h[kk] = (bf16)f4c(rg[kk], j);
        pu.h[kk] = (bf16)f4c(ru[kk], j);
      }
      *(uint4*)&Bgs[buf][(nB + j) * LDS_STRIDE + kB] = pg.u;
      *(uint4*)&Bus[buf][(nB + j) * LDS_STRIDE + kB] = pu.u;
    }
  };

  // ---- prologue: stage 0 ----
  fillA(0, 0);
  loadB(0);
  storeB(0);
  async_wait_all();
  __syncthreads();

  const int S = K / BK;
  for (int s = 0; s < S; ++s) {
    const int cur = s & 1;
    const int nxt = cur ^ 1;
    if (s + 1 < S) {                 // issue next stage traffic first
      fillA(nxt, (s + 1) * BK);
      loadB((s + 1) * BK);
    }
    // compute current stage (hides global latency of next stage)
#pragma unroll
    for (int ks = 0; ks < BK; ks += 32) {
      v16bf a[4], bg[2], bu[2];
#pragma unroll
      for (int fm = 0; fm < 4; ++fm) a[fm] = load_a_frag(As[cur], wm + fm * 16, ks, lane);
#pragma unroll
      for (int fn = 0; fn < 2; ++fn) {
        bg[fn] = load_b_frag(Bgs[cur], wn + fn * 16, ks, lane);
        bu[fn] = load_b_frag(Bus[cur], wn + fn * 16, ks, lane);
      }
#pragma unroll
      for (int fm = 0; fm < 4; ++fm)
#pragma unroll
        for (int fn = 0; fn < 2; ++fn) {
          cg[fm][fn] = __builtin_amdgcn_wmma_f32_16x16x32_bf16(
              false, a[fm], false, bg[fn], (short)0, cg[fm][fn], false, false);
          cu[fm][fn] = __builtin_amdgcn_wmma_f32_16x16x32_bf16(
              false, a[fm], false, bu[fn], (short)0, cu[fm][fn], false, false);
        }
    }
    if (s + 1 < S) storeB(nxt);      // convert+store after compute (loads done)
    async_wait_all();
    __syncthreads();
  }

  // ---- epilogue: SwiGLU, store bf16 activations ----
  const int rhi   = (lane & 16) ? 8 : 0;
  const int ncol0 = nbase + wn + (lane & 15);
#pragma unroll
  for (int fm = 0; fm < 4; ++fm)
#pragma unroll
    for (int fn = 0; fn < 2; ++fn)
#pragma unroll
      for (int r = 0; r < 8; ++r) {
        int pos = mbase + wm + fm * 16 + rhi + r;
        if (pos < cnt) {
          float g = cg[fm][fn][r];
          float u = cu[fm][fn][r];
          float s = g / (1.0f + __expf(-g)) * u;   // silu(g) * u
          Act[(size_t)(rowOff + pos) * N + (ncol0 + fn * 16)] = (bf16)s;
        }
      }
}

// ---------------------------------------------------------------------------
// GEMM2: down-projection. shared mode: Out = Act @ Wd (plain store, inits y).
// routed mode: Out[token] += weight * (Act @ Wd_e) via fp32 HW atomics.
// N fixed to H_DIM; K = IS_DIM (shared) or I_DIM (routed); ldA = K.
// Same double-buffered / async-copy pipeline as GEMM1.
// ---------------------------------------------------------------------------
__global__ __launch_bounds__(256)
void moe_gemm2_down(const bf16* __restrict__ Act,  // [rows][K] bf16
                    const float* __restrict__ Wd,  // [K][H_DIM] fp32 (per expert K*H)
                    float* __restrict__ Out,       // [T][H_DIM] fp32
                    int K,
                    const int* __restrict__ counts,
                    const int* __restrict__ offsets,
                    const int* __restrict__ elist,
                    const float* __restrict__ wlist) {
  __shared__ bf16  As[2][TILE_ELEMS];
  __shared__ bf16  Bs[2][TILE_ELEMS];
  __shared__ int   tokLds[128];
  __shared__ float wLds[128];

  const int tid  = threadIdx.x;
  const int lane = tid & 31;
  const int wave = tid >> 5;
  const int mtile = blockIdx.x, ntile = blockIdx.y, e = blockIdx.z;
  const bool routed = (counts != nullptr);

  int cnt = T_NUM, rowOff = 0;
  if (routed) {
    cnt = counts[e];
    if (mtile * 128 >= cnt) return;
    rowOff = offsets[e];
    Wd += (size_t)e * K * H_DIM;
  }
  const int mbase = mtile * 128;
  const int nbase = ntile * 128;

  if (routed && tid < 128) {
    int pos = mbase + tid;
    tokLds[tid] = (pos < cnt) ? elist[(size_t)e * T_NUM + pos] : 0;
    wLds[tid]   = (pos < cnt) ? wlist[(size_t)e * T_NUM + pos] : 0.0f;
  }
  __syncthreads();

  const int wm = (wave & 1) * 64;
  const int wn = (wave >> 1) * 32;
  const int nB = (tid & 31) * 4;
  const int kB = (tid >> 5) * 8;

  float4 rw[8];

  v8f c[4][2];
#pragma unroll
  for (int i = 0; i < 4; ++i)
#pragma unroll
    for (int j = 0; j < 2; ++j) c[i][j] = {};

  auto fillA = [&](int buf, int kb) {
#pragma unroll
    for (int i = 0; i < 4; ++i) {
      int idx = tid + i * 256;
      int r = idx >> 3;
      int cc = (idx & 7) * 8;
      int pos = mbase + r;
      int prow = (pos < cnt) ? pos : (cnt - 1);      // clamp (masked at store)
      async_copy16(Act + (size_t)(rowOff + prow) * K + kb + cc,
                   &As[buf][r * LDS_STRIDE + cc]);
    }
  };
  auto loadB = [&](int kb) {
#pragma unroll
    for (int kk = 0; kk < 8; ++kk)
      rw[kk] = *(const float4*)(Wd + (size_t)(kb + kB + kk) * H_DIM + nbase + nB);
  };
  auto storeB = [&](int buf) {
#pragma unroll
    for (int j = 0; j < 4; ++j) {
      Pack8 pw;
#pragma unroll
      for (int kk = 0; kk < 8; ++kk) pw.h[kk] = (bf16)f4c(rw[kk], j);
      *(uint4*)&Bs[buf][(nB + j) * LDS_STRIDE + kB] = pw.u;
    }
  };

  fillA(0, 0);
  loadB(0);
  storeB(0);
  async_wait_all();
  __syncthreads();

  const int S = K / BK;
  for (int s = 0; s < S; ++s) {
    const int cur = s & 1;
    const int nxt = cur ^ 1;
    if (s + 1 < S) {
      fillA(nxt, (s + 1) * BK);
      loadB((s + 1) * BK);
    }
#pragma unroll
    for (int ks = 0; ks < BK; ks += 32) {
      v16bf a[4], b[2];
#pragma unroll
      for (int fm = 0; fm < 4; ++fm) a[fm] = load_a_frag(As[cur], wm + fm * 16, ks, lane);
#pragma unroll
      for (int fn = 0; fn < 2; ++fn) b[fn] = load_b_frag(Bs[cur], wn + fn * 16, ks, lane);
#pragma unroll
      for (int fm = 0; fm < 4; ++fm)
#pragma unroll
        for (int fn = 0; fn < 2; ++fn)
          c[fm][fn] = __builtin_amdgcn_wmma_f32_16x16x32_bf16(
              false, a[fm], false, b[fn], (short)0, c[fm][fn], false, false);
    }
    if (s + 1 < S) storeB(nxt);
    async_wait_all();
    __syncthreads();
  }

  const int rhi   = (lane & 16) ? 8 : 0;
  const int ncol0 = nbase + wn + (lane & 15);
#pragma unroll
  for (int fm = 0; fm < 4; ++fm)
#pragma unroll
    for (int r = 0; r < 8; ++r) {
      int lr  = wm + fm * 16 + rhi + r;
      int pos = mbase + lr;
      if (pos < cnt) {
        if (routed) {
          int tok = tokLds[lr];
          float w = wLds[lr];
#pragma unroll
          for (int fn = 0; fn < 2; ++fn)
            unsafeAtomicAdd(&Out[(size_t)tok * H_DIM + ncol0 + fn * 16],
                            w * c[fm][fn][r]);
        } else {
#pragma unroll
          for (int fn = 0; fn < 2; ++fn)
            Out[(size_t)pos * H_DIM + ncol0 + fn * 16] = c[fm][fn][r];
        }
      }
    }
}

// ---------------------------------------------------------------------------
extern "C" void kernel_launch(void* const* d_in, const int* in_sizes, int n_in,
                              void* d_out, int out_size, void* d_ws, size_t ws_size,
                              hipStream_t stream) {
  const float* x       = (const float*)d_in[0];  // [T][H]
  const float* gate_w  = (const float*)d_in[1];  // [E][H]
  const float* we_gate = (const float*)d_in[2];  // [E][H][I]
  const float* we_up   = (const float*)d_in[3];  // [E][H][I]
  const float* we_down = (const float*)d_in[4];  // [E][I][H]
  const float* ws_gate = (const float*)d_in[5];  // [H][IS]
  const float* ws_up   = (const float*)d_in[6];  // [H][IS]
  const float* ws_down = (const float*)d_in[7];  // [IS][H]
  float* out = (float*)d_out;

  // workspace carve (~80.2 MB total)
  char* p = (char*)d_ws;
  bf16*  xb      = (bf16*)p;  p += (size_t)T_NUM * H_DIM * sizeof(bf16);   // 32 MiB
  bf16*  act     = (bf16*)p;  p += (size_t)T_NUM * IS_DIM * sizeof(bf16);  // 44 MiB (== 2T*I)
  int*   elist   = (int*)p;   p += (size_t)E_NUM * T_NUM * sizeof(int);
  float* wlist   = (float*)p; p += (size_t)E_NUM * T_NUM * sizeof(float);
  int*   counts  = (int*)p;   p += 256;
  int*   offsets = (int*)p;

  (void)hipMemsetAsync(counts, 0, E_NUM * sizeof(int), stream);

  // 1) x -> bf16
  moe_convert_x<<<(T_NUM * H_DIM) / (256 * 8), 256, 0, stream>>>(x, xb);

  // 2) gate + routing lists
  moe_gate_route<<<T_NUM / 8, 256, 0, stream>>>(x, gate_w, counts, elist, wlist);

  // 3) compact offsets
  moe_prefix8<<<1, 32, 0, stream>>>(counts, offsets);

  // 4) shared expert: act = swiglu(x @ ws_gate, x @ ws_up)   [T x IS]
  moe_gemm1_swiglu<<<dim3(T_NUM / 128, IS_DIM / 128, 1), 256, 0, stream>>>(
      xb, ws_gate, ws_up, act, H_DIM, IS_DIM, nullptr, nullptr, nullptr);

  // 5) shared expert: out = act @ ws_down  (initializes d_out)
  moe_gemm2_down<<<dim3(T_NUM / 128, H_DIM / 128, 1), 256, 0, stream>>>(
      act, ws_down, out, IS_DIM, nullptr, nullptr, nullptr, nullptr);

  // 6) routed experts: act = swiglu(gather(x) @ we_gate_e, ... @ we_up_e)  [2T x I]
  moe_gemm1_swiglu<<<dim3(T_NUM / 128, I_DIM / 128, E_NUM), 256, 0, stream>>>(
      xb, we_gate, we_up, act, H_DIM, I_DIM, counts, offsets, elist);

  // 7) routed experts: out[token] += w * (act @ we_down_e)
  moe_gemm2_down<<<dim3(T_NUM / 128, H_DIM / 128, E_NUM), 256, 0, stream>>>(
      act, we_down, out, I_DIM, counts, offsets, elist, wlist);
}
